// MultiHeadAttention_13511967113606
// MI455X (gfx1250) — matrix operand order repaired and checked by reading, not compiled, not run on hardware
//
#include <hip/hip_runtime.h>
#include <hip/hip_bf16.h>

typedef __attribute__((ext_vector_type(16))) _Float16 v16h;
typedef __attribute__((ext_vector_type(8)))  _Float16 h8;
typedef __attribute__((ext_vector_type(8)))  float    v8f;

#define S_LEN   4096
#define DMODEL  512
#define NHEADS  8
#define DK      64
#define BATCH   2
#define BSROWS  (BATCH * S_LEN)   // 8192 rows of the flattened activations

static __device__ __forceinline__ v16h make_frag(const _Float16* lo, const _Float16* hi) {
    h8 a = *(const h8*)lo;
    h8 b = *(const h8*)hi;
    v16h r;
#pragma unroll
    for (int i = 0; i < 8; ++i) { r[i] = a[i]; r[i + 8] = b[i]; }
    return r;
}

static __device__ __forceinline__ v8f wmma16(v16h a, v16h b, v8f c) {
    return __builtin_amdgcn_wmma_f32_16x16x32_f16(false, a, false, b, (short)0, c, false, false);
}

static __device__ __forceinline__ v8f v8f_zero() {
    v8f z = {0.f, 0.f, 0.f, 0.f, 0.f, 0.f, 0.f, 0.f};
    return z;
}

// Max-reduce across the 16-lane DPP row (both half-waves independently).
// Sequence: xor1, xor2, then mirrors which complete the butterfly once the
// smaller groups are uniform.
static __device__ __forceinline__ float dpp_rowmax16(float x) {
    float y;
    y = __int_as_float(__builtin_amdgcn_update_dpp(
            0, __float_as_int(x), 0xB1 /*quad_perm(1,0,3,2)*/, 0xF, 0xF, true));
    x = fmaxf(x, y);
    y = __int_as_float(__builtin_amdgcn_update_dpp(
            0, __float_as_int(x), 0x4E /*quad_perm(2,3,0,1)*/, 0xF, 0xF, true));
    x = fmaxf(x, y);
    y = __int_as_float(__builtin_amdgcn_update_dpp(
            0, __float_as_int(x), 0x141 /*row_half_mirror*/, 0xF, 0xF, true));
    x = fmaxf(x, y);
    y = __int_as_float(__builtin_amdgcn_update_dpp(
            0, __float_as_int(x), 0x140 /*row_mirror*/, 0xF, 0xF, true));
    x = fmaxf(x, y);
    return x;
}

// ---------------------------------------------------------------- converters
__global__ void cvt_f32_to_f16(const float* __restrict__ src,
                               _Float16* __restrict__ dst, int n) {
    int i = (blockIdx.x * blockDim.x + threadIdx.x) * 4;
    if (i + 3 < n) {
        float4 v = *(const float4*)(src + i);
        dst[i + 0] = (_Float16)v.x;
        dst[i + 1] = (_Float16)v.y;
        dst[i + 2] = (_Float16)v.z;
        dst[i + 3] = (_Float16)v.w;
    }
}

// Wt[k][n] = (f16) W[n][k]   (so GEMM computes x @ W^T with contiguous B-frags)
__global__ void transpose_w_f16(const float* __restrict__ w,
                                _Float16* __restrict__ wt) {
    int idx = blockIdx.x * blockDim.x + threadIdx.x;   // idx = k*512 + n
    int k = idx >> 9;
    int n = idx & 511;
    wt[idx] = (_Float16)w[n * DMODEL + k];
}

// ---------------------------------------------------------------- WMMA GEMM
// y[m,n] = sum_k X[m,k] * Wt[k,n] + bias[n]; each wave owns a 16x64 tile.
// OUTF32=0: write f16 head-major [B,H,S,DK] (outHM) and/or transposed
//           [B,H,DK,S] (outT), with scale folded in.
// OUTF32=1: write f32 row-major [BSROWS, DMODEL] (outF32).
template <int OUTF32>
__global__ void __launch_bounds__(256) gemm_wmma(
    const _Float16* __restrict__ X, const _Float16* __restrict__ Wt,
    const float* __restrict__ bias, float scale,
    _Float16* __restrict__ outHM, _Float16* __restrict__ outT,
    float* __restrict__ outF32) {
    const int lane = threadIdx.x & 31;
    const int wid  = (blockIdx.x * blockDim.x + threadIdx.x) >> 5;
    const int m0   = (wid >> 3) * 16;   // 512 row tiles
    const int n0   = (wid & 7) * 64;    // 8 column tiles of 64
    const int lm   = lane & 15;
    const int hh   = lane >> 4;

    v8f acc[4];
#pragma unroll
    for (int c = 0; c < 4; ++c) acc[c] = v8f_zero();

    const _Float16* xrow = X + (size_t)(m0 + lm) * DMODEL;
    for (int kk = 0; kk < DMODEL; kk += 32) {
        v16h a = make_frag(xrow + kk + 8 * hh, xrow + kk + 16 + 8 * hh);
        const _Float16* wrow = Wt + (size_t)kk * DMODEL + n0 + lm;
#pragma unroll
        for (int c = 0; c < 4; ++c) {
            v16h b;
            for (int s = 0; s < 16; ++s) b[s] = wrow[(size_t)((s & 7) + 8 * hh + 16 * (s >> 3)) * DMODEL + 16 * c];
            acc[c] = wmma16(a, b, acc[c]);
        }
    }

#pragma unroll
    for (int c = 0; c < 4; ++c) {
        const int n  = n0 + 16 * c + lm;     // C-layout: column lives on lane
        const float bv = bias[n];
#pragma unroll
        for (int r = 0; r < 8; ++r) {
            float y = (acc[c][r] + bv) * scale;
            int g = m0 + r + 8 * hh;         // C-layout: row = vgpr + 8*half
            if (OUTF32) {
                outF32[(size_t)g * DMODEL + n] = y;
            } else {
                int bi = g >> 12, s = g & (S_LEN - 1);
                int hd = n >> 6,  dk = n & 63;
                if (outHM)
                    outHM[(((size_t)(bi * NHEADS + hd)) * S_LEN + s) * DK + dk] =
                        (_Float16)y;
                if (outT)
                    outT[(((size_t)(bi * NHEADS + hd)) * DK + dk) * S_LEN + s] =
                        (_Float16)y;
            }
        }
    }
}

// ---------------------------------------------------------------- flash attn
// One wave = 16 query rows of one (b,h). Streams keys 32 at a time.
__global__ void __launch_bounds__(256) flash_attn(
    const _Float16* __restrict__ Qh,   // [B,H,S,DK], pre-scaled by 1/8
    const _Float16* __restrict__ KhT,  // [B,H,DK,S]
    const _Float16* __restrict__ Vh,   // [B,H,S,DK]
    const int* __restrict__ mask,      // [B,S]
    _Float16* __restrict__ attnO) {    // [BSROWS, DMODEL] f16
    __shared__ _Float16 plds[8 * 512];   // 16x32 f16 P-tile per wave
    const int lane = threadIdx.x & 31;
    const int lwid = threadIdx.x >> 5;
    const int wid  = blockIdx.x * 8 + lwid;
    const int qt   = wid & 255;          // 256 q-tiles per (b,h)
    const int bh   = wid >> 8;           // 0..15
    const int b    = bh >> 3, hd = bh & 7;
    const int q0   = qt * 16;
    const int lm   = lane & 15, hh = lane >> 4;

    const _Float16* qb  = Qh  + (size_t)bh * S_LEN * DK;
    const _Float16* ktb = KhT + (size_t)bh * DK * S_LEN;
    const _Float16* vb  = Vh  + (size_t)bh * S_LEN * DK;
    const int* mrow = mask + b * S_LEN;
    _Float16* lb = plds + lwid * 512;

    // Q fragments: rows m = q0+lm, dk split into two K=32 chunks
    const _Float16* qrow = qb + (size_t)(q0 + lm) * DK;
    const v16h aq0 = make_frag(qrow + 8 * hh,      qrow + 16 + 8 * hh);
    const v16h aq1 = make_frag(qrow + 32 + 8 * hh, qrow + 48 + 8 * hh);

    // All-ones B fragment: wmma(ap, ones) -> per-row sums of P in C-layout
    v16h ones;
#pragma unroll
    for (int i = 0; i < 16; ++i) ones[i] = (_Float16)1.0f;

    v8f o[4];
    float mrun[8], lrun[8];
#pragma unroll
    for (int c = 0; c < 4; ++c) o[c] = v8f_zero();
#pragma unroll
    for (int r = 0; r < 8; ++r) { mrun[r] = -1e30f; lrun[r] = 0.f; }

    for (int key0 = 0; key0 < S_LEN; key0 += 32) {
        const _Float16* kt0  = ktb + key0 + lm;                          // dk 0..31
        const _Float16* kt1  = ktb + (size_t)32 * S_LEN + key0 + lm;     // dk 32..63
        const _Float16* vrow = vb + (size_t)key0 * DK + lm;

        // keep the HBM/L2 stream ahead of the matrix pipe
        __builtin_prefetch(kt0 + 64, 0, 3);
        __builtin_prefetch(kt1 + 64, 0, 3);
        __builtin_prefetch(vrow + 32 * DK, 0, 3);

        // ---- scores: two 16-key tiles, dk=64 via two K=32 WMMAs each
        v8f s0 = v8f_zero(), s1 = v8f_zero();
        {
            v16h b00, b01, b10, b11;
            for (int s = 0; s < 16; ++s) {
                size_t ko = (size_t)((s & 7) + 8 * hh + 16 * (s >> 3)) * S_LEN;
                b00[s] = kt0[ko]; b01[s] = kt1[ko];
                b10[s] = kt0[ko + 16]; b11[s] = kt1[ko + 16];
            }
            s0 = wmma16(aq0, b00, s0);
            s0 = wmma16(aq1, b01, s0);
            s1 = wmma16(aq0, b10, s1);
            s1 = wmma16(aq1, b11, s1);
        }
        // ---- mask (column = key lives on lane)
        const float mb0 = mrow[key0 + lm]      ? 0.f : -1e9f;
        const float mb1 = mrow[key0 + 16 + lm] ? 0.f : -1e9f;
#pragma unroll
        for (int r = 0; r < 8; ++r) { s0[r] += mb0; s1[r] += mb1; }

        // ---- online softmax: row max via DPP16 butterflies (no LDS traffic)
        float p0[8], p1[8], alpha[8];
#pragma unroll
        for (int r = 0; r < 8; ++r) {
            float t = dpp_rowmax16(fmaxf(s0[r], s1[r]));
            float mnew = fmaxf(mrun[r], t);
            alpha[r] = __expf(mrun[r] - mnew);
            mrun[r]  = mnew;
            p0[r] = __expf(s0[r] - mnew);
            p1[r] = __expf(s1[r] - mnew);
        }

        // ---- transpose P (C-layout) -> A-layout via per-wave LDS tile
#pragma unroll
        for (int r = 0; r < 8; ++r) {
            lb[(r + 8 * hh) * 32 + lm]      = (_Float16)p0[r];
            lb[(r + 8 * hh) * 32 + 16 + lm] = (_Float16)p1[r];
        }
        asm volatile("s_wait_dscnt 0" ::: "memory");
        v16h ap = make_frag(lb + lm * 32 + 8 * hh, lb + lm * 32 + 16 + 8 * hh);

        // ---- row sums of P via the matrix pipe (C-layout broadcast)
        v8f lsum = wmma16(ap, ones, v8f_zero());
#pragma unroll
        for (int r = 0; r < 8; ++r)
            lrun[r] = lrun[r] * alpha[r] + lsum[r];

        // ---- P @ V : B-frag lane = dk column, 4 chunks of 16 dk columns
#pragma unroll
        for (int c = 0; c < 4; ++c) {
            v16h bvf;
            for (int s = 0; s < 16; ++s) bvf[s] = vrow[(size_t)((s & 7) + 8 * hh + 16 * (s >> 3)) * DK + 16 * c];
#pragma unroll
            for (int r = 0; r < 8; ++r) o[c][r] *= alpha[r];
            o[c] = wmma16(ap, bvf, o[c]);
        }
    }

    // ---- normalize and write f16 [row, h*64+dk]
    float rinv[8];
#pragma unroll
    for (int r = 0; r < 8; ++r) rinv[r] = 1.0f / lrun[r];
#pragma unroll
    for (int c = 0; c < 4; ++c) {
#pragma unroll
        for (int r = 0; r < 8; ++r) {
            int g = b * S_LEN + q0 + r + 8 * hh;
            int n = hd * DK + 16 * c + lm;
            attnO[(size_t)g * DMODEL + n] = (_Float16)(o[c][r] * rinv[r]);
        }
    }
}

// ---------------------------------------------------------------- launcher
extern "C" void kernel_launch(void* const* d_in, const int* in_sizes, int n_in,
                              void* d_out, int out_size, void* d_ws, size_t ws_size,
                              hipStream_t stream) {
    (void)in_sizes; (void)n_in; (void)out_size; (void)ws_size;
    const float* q  = (const float*)d_in[0];
    const float* k  = (const float*)d_in[1];
    const float* v  = (const float*)d_in[2];
    const int*  msk = (const int*)  d_in[3];
    const float* wq = (const float*)d_in[4];  const float* bq = (const float*)d_in[5];
    const float* wk = (const float*)d_in[6];  const float* bk = (const float*)d_in[7];
    const float* wv = (const float*)d_in[8];  const float* bv = (const float*)d_in[9];
    const float* wo = (const float*)d_in[10]; const float* bo = (const float*)d_in[11];
    float* out = (float*)d_out;

    char* ws = (char*)d_ws;
    size_t off = 0;
    auto alloc = [&](size_t bytes) -> void* {
        void* p = ws + off;
        off += (bytes + 255) & ~(size_t)255;
        return p;
    };
    const size_t actH = (size_t)BSROWS * DMODEL * sizeof(_Float16);   // 8 MB
    const size_t wH   = (size_t)DMODEL * DMODEL * sizeof(_Float16);   // 0.5 MB
    const size_t headH = (size_t)BATCH * NHEADS * S_LEN * DK * sizeof(_Float16); // 8 MB

    _Float16* xq  = (_Float16*)alloc(actH);
    _Float16* xk  = (_Float16*)alloc(actH);
    _Float16* xv  = (_Float16*)alloc(actH);
    _Float16* wtq = (_Float16*)alloc(wH);
    _Float16* wtk = (_Float16*)alloc(wH);
    _Float16* wtv = (_Float16*)alloc(wH);
    _Float16* wto = (_Float16*)alloc(wH);
    _Float16* Qh  = (_Float16*)alloc(headH);
    _Float16* KhT = (_Float16*)alloc(headH);
    _Float16* Vh  = (_Float16*)alloc(headH);
    _Float16* attnO = xq;   // xq is dead after the Q projection

    const int THR = 256;
    const int nAct = BSROWS * DMODEL;          // 4M elements
    cvt_f32_to_f16<<<nAct / 4 / THR, THR, 0, stream>>>(q, xq, nAct);
    cvt_f32_to_f16<<<nAct / 4 / THR, THR, 0, stream>>>(k, xk, nAct);
    cvt_f32_to_f16<<<nAct / 4 / THR, THR, 0, stream>>>(v, xv, nAct);

    const int nW = DMODEL * DMODEL;            // 256K
    transpose_w_f16<<<nW / THR, THR, 0, stream>>>(wq, wtq);
    transpose_w_f16<<<nW / THR, THR, 0, stream>>>(wk, wtk);
    transpose_w_f16<<<nW / THR, THR, 0, stream>>>(wv, wtv);
    transpose_w_f16<<<nW / THR, THR, 0, stream>>>(wo, wto);

    // 4096 wave tiles / 8 waves per block = 512 blocks
    gemm_wmma<0><<<512, THR, 0, stream>>>(xq, wtq, bq, 0.125f, Qh, nullptr, nullptr);
    gemm_wmma<0><<<512, THR, 0, stream>>>(xk, wtk, bk, 1.0f, nullptr, KhT, nullptr);
    gemm_wmma<0><<<512, THR, 0, stream>>>(xv, wtv, bv, 1.0f, Vh, nullptr, nullptr);

    flash_attn<<<512, THR, 0, stream>>>(Qh, KhT, Vh, msk, attnO);

    gemm_wmma<1><<<512, THR, 0, stream>>>(attnO, wto, bo, 1.0f, nullptr, nullptr, out);
}
